// LocalFeatureTransformer_71983651881332
// MI455X (gfx1250) — matrix operand order repaired and checked by reading, not compiled, not run on hardware
//
#include <hip/hip_runtime.h>
#include <hip/hip_bf16.h>
#include <math.h>

typedef __attribute__((ext_vector_type(16))) _Float16 v16h;
typedef __attribute__((ext_vector_type(8)))  _Float16 v8h;
typedef __attribute__((ext_vector_type(8)))  float    v8f;
typedef __attribute__((ext_vector_type(4)))  int      v4i;

#define D_MODEL   256
#define NHEAD     8
#define HDIM      32
#define SEQ       4800
#define BATCH     2
#define MROWS     (BATCH * SEQ)                 // 9600
#define FEAT_ELEMS ((size_t)MROWS * D_MODEL)    // 2,457,600

#define EPI_NONE 0
#define EPI_RELU 1
#define EPI_ELU1 2

#ifndef __has_builtin
#define __has_builtin(x) 0
#endif

#if __has_builtin(__builtin_amdgcn_global_load_async_to_lds_b128)
#define HAVE_ASYNC 1
#else
#define HAVE_ASYNC 0
#endif

#if HAVE_ASYNC
typedef v4i __attribute__((address_space(1)))* g1v4i;   // global (AS1) int4*
typedef v4i __attribute__((address_space(3)))* l3v4i;   // LDS (AS3) int4*
__device__ __forceinline__ void async_cp16(void* lds, const void* g) {
  // 16B per lane, memory -> LDS, tracked by ASYNCcnt
  __builtin_amdgcn_global_load_async_to_lds_b128((g1v4i)g, (l3v4i)lds, 0, 0);
}
#if __has_builtin(__builtin_amdgcn_s_wait_asynccnt)
#define WAIT_ASYNC_3() __builtin_amdgcn_s_wait_asynccnt(3)
#define WAIT_ASYNC_0() __builtin_amdgcn_s_wait_asynccnt(0)
#else
#define WAIT_ASYNC_3() asm volatile("s_wait_asynccnt 0x3" ::: "memory")
#define WAIT_ASYNC_0() asm volatile("s_wait_asynccnt 0x0" ::: "memory")
#endif
#endif

template<int EPI>
__device__ __forceinline__ float epi_apply(float x) {
  if (EPI == EPI_RELU) return x > 0.f ? x : 0.f;
  if (EPI == EPI_ELU1) return x > 0.f ? x + 1.f : __expf(x); // elu(x)+1
  return x;
}

// ---------------------------------------------------------------------------
// WMMA GEMM, all-f16 operands: C = epi( A[M x K] @ W[K x N] ).
// A is f16 row-major (optionally split A0(K0 cols) | A1(K1 cols) for concat-MLP).
// WT is f16 pre-transposed [N x K] so both LDS tiles stage as pure 16B copies.
// Block tile 128(M) x 64(N), 8 waves, wave -> 32x32 = 2x2 v_wmma_f32_16x16x32_f16.
// Double-buffered LDS; async-to-LDS staging when the builtin is available.
// ---------------------------------------------------------------------------
template<int EPI, bool OUTH>
__global__ __launch_bounds__(256)
void gemm_wmma_f16(const _Float16* __restrict__ A0, const _Float16* __restrict__ A1,
                   int K0, int K1,
                   const _Float16* __restrict__ WT, int K, int N,
                   float* __restrict__ Cf, _Float16* __restrict__ Ch)
{
  __shared__ _Float16 As[2][128][40];   // 128 rows x 32 k, padded rows (16B aligned)
  __shared__ _Float16 Bs[2][64][40];    // 64 n x 32 k

  const int tid   = threadIdx.x;
  const int lane  = tid & 31;
  const int wid   = tid >> 5;
  const int waveM = wid & 3;
  const int waveN = wid >> 2;
  const int rowB  = blockIdx.y * 128;
  const int colB  = blockIdx.x * 64;

  v8f c00 = {}; v8f c01 = {}; v8f c10 = {}; v8f c11 = {};

  const int ar    = lane & 15;
  const int koff  = (lane >> 4) * 8;    // A frag: lanes16-31 hold K+8 block
  const int bkoff = (lane >> 4) * 8;    // B frag: lanes16-31 hold K+8 block
  const int bc    = lane & 15;

  auto compute = [&](int buf) {
    v16h a0, a1, b0, b1;
    {
      int r = waveM * 32 + ar;
      v8h lo = *(const v8h*)(&As[buf][r][koff]);
      v8h hi = *(const v8h*)(&As[buf][r][16 + koff]);
#pragma unroll
      for (int i = 0; i < 8; ++i) { a0[i] = lo[i]; a0[8 + i] = hi[i]; }
      r += 16;
      lo = *(const v8h*)(&As[buf][r][koff]);
      hi = *(const v8h*)(&As[buf][r][16 + koff]);
#pragma unroll
      for (int i = 0; i < 8; ++i) { a1[i] = lo[i]; a1[8 + i] = hi[i]; }
    }
    {
      int c = waveN * 32 + bc;
      v8h lo = *(const v8h*)(&Bs[buf][c][bkoff]);
      v8h hi = *(const v8h*)(&Bs[buf][c][16 + bkoff]);
#pragma unroll
      for (int i = 0; i < 8; ++i) { b0[i] = lo[i]; b0[8 + i] = hi[i]; }
      c += 16;
      lo = *(const v8h*)(&Bs[buf][c][bkoff]);
      hi = *(const v8h*)(&Bs[buf][c][16 + bkoff]);
#pragma unroll
      for (int i = 0; i < 8; ++i) { b1[i] = lo[i]; b1[8 + i] = hi[i]; }
    }
    c00 = __builtin_amdgcn_wmma_f32_16x16x32_f16(false, a0, false, b0, (short)0, c00, false, false);
    c01 = __builtin_amdgcn_wmma_f32_16x16x32_f16(false, a0, false, b1, (short)0, c01, false, false);
    c10 = __builtin_amdgcn_wmma_f32_16x16x32_f16(false, a1, false, b0, (short)0, c10, false, false);
    c11 = __builtin_amdgcn_wmma_f32_16x16x32_f16(false, a1, false, b1, (short)0, c11, false, false);
  };

#if HAVE_ASYNC
  auto issue_tile = [&](int buf, int kb) {
    const _Float16* Abase = (kb < K0) ? A0 + kb : A1 + (kb - K0);
    const int lda = (kb < K0) ? K0 : K1;
#pragma unroll
    for (int it = 0; it < 2; ++it) {               // A tile: 512 x 16B chunks
      int idx = tid + it * 256;
      int r = idx >> 2, c8 = (idx & 3) * 8;
      async_cp16(&As[buf][r][c8], Abase + (size_t)(rowB + r) * lda + c8);
    }
    {                                              // B tile: 256 x 16B chunks
      int n = tid >> 2, c8 = (tid & 3) * 8;
      async_cp16(&Bs[buf][n][c8], WT + (size_t)(colB + n) * K + kb + c8);
    }
  };

  const int KT = K >> 5;
  issue_tile(0, 0);
  for (int kt = 0; kt < KT; ++kt) {
    const int buf = kt & 1;
    if (kt + 1 < KT) { issue_tile(buf ^ 1, (kt + 1) << 5); WAIT_ASYNC_3(); }
    else             { WAIT_ASYNC_0(); }
    __syncthreads();          // all waves' async fills for this tile complete
    compute(buf);
    __syncthreads();          // safe to overwrite the other buffer next iter
  }
#else
  for (int kb = 0; kb < K; kb += 32) {
    const _Float16* Abase = (kb < K0) ? A0 + kb : A1 + (kb - K0);
    const int lda = (kb < K0) ? K0 : K1;
#pragma unroll
    for (int it = 0; it < 2; ++it) {
      int idx = tid + it * 256;
      int r = idx >> 2, c8 = (idx & 3) * 8;
      *(v8h*)(&As[0][r][c8]) = *(const v8h*)(Abase + (size_t)(rowB + r) * lda + c8);
    }
    {
      int n = tid >> 2, c8 = (tid & 3) * 8;
      *(v8h*)(&Bs[0][n][c8]) = *(const v8h*)(WT + (size_t)(colB + n) * K + kb + c8);
    }
    __syncthreads();
    compute(0);
    __syncthreads();
  }
#endif

  // C/D layout: row = fragbase + 8*(lane>=16) + i, col = lane&15
  const int crow = rowB + waveM * 32 + (lane >> 4) * 8;
  const int ccol = colB + waveN * 32 + (lane & 15);
#pragma unroll
  for (int i = 0; i < 8; ++i) {
    if (OUTH) {
      Ch[(size_t)(crow + i)      * N + ccol]      = (_Float16)epi_apply<EPI>(c00[i]);
      Ch[(size_t)(crow + i)      * N + ccol + 16] = (_Float16)epi_apply<EPI>(c01[i]);
      Ch[(size_t)(crow + 16 + i) * N + ccol]      = (_Float16)epi_apply<EPI>(c10[i]);
      Ch[(size_t)(crow + 16 + i) * N + ccol + 16] = (_Float16)epi_apply<EPI>(c11[i]);
    } else {
      Cf[(size_t)(crow + i)      * N + ccol]      = epi_apply<EPI>(c00[i]);
      Cf[(size_t)(crow + i)      * N + ccol + 16] = epi_apply<EPI>(c01[i]);
      Cf[(size_t)(crow + 16 + i) * N + ccol]      = epi_apply<EPI>(c10[i]);
      Cf[(size_t)(crow + 16 + i) * N + ccol + 16] = epi_apply<EPI>(c11[i]);
    }
  }
}

// ---------------------------------------------------------------------------
// KV[b,h,d,v] = sum_s sigmaK[b,s,h,d] * V[b,s,h,v]; Ksum[b,h,d] = sum_s sigmaK.
// (the 1/s_len on v and *s_len on out cancel exactly, so both are dropped)
// ---------------------------------------------------------------------------
__global__ __launch_bounds__(256)
void kv_reduce_kernel(const float* __restrict__ sigK, const float* __restrict__ V,
                      float* __restrict__ KV, float* __restrict__ Ksum)
{
  __shared__ float Ks[64][32];
  __shared__ float Vs[64][33];

  const int bh = blockIdx.x;
  const int b  = bh >> 3, h = bh & 7;
  const int tid = threadIdx.x;
  const int d   = tid >> 3;
  const int v4  = (tid & 7) * 4;

  float a0 = 0.f, a1 = 0.f, a2 = 0.f, a3 = 0.f, ks = 0.f;
  const size_t rowBase = (size_t)b * SEQ;

  for (int s0 = 0; s0 < SEQ; s0 += 64) {
#pragma unroll
    for (int j = 0; j < 8; ++j) {
      int idx = tid + j * 256;
      int r = idx >> 5, c = idx & 31;
      size_t off = (rowBase + s0 + r) * D_MODEL + (size_t)h * HDIM + c;
      Ks[r][c] = sigK[off];
      Vs[r][c] = V[off];
    }
    __syncthreads();
#pragma unroll 4
    for (int r = 0; r < 64; ++r) {
      float kd = Ks[r][d];
      a0 += kd * Vs[r][v4 + 0];
      a1 += kd * Vs[r][v4 + 1];
      a2 += kd * Vs[r][v4 + 2];
      a3 += kd * Vs[r][v4 + 3];
    }
    if (tid < 32) {
#pragma unroll 4
      for (int r = 0; r < 64; ++r) ks += Ks[r][tid];
    }
    __syncthreads();
  }
  float* kvo = KV + (size_t)bh * (HDIM * HDIM);
  kvo[d * 32 + v4 + 0] = a0;
  kvo[d * 32 + v4 + 1] = a1;
  kvo[d * 32 + v4 + 2] = a2;
  kvo[d * 32 + v4 + 3] = a3;
  if (tid < 32) Ksum[bh * HDIM + tid] = ks;
}

// ---------------------------------------------------------------------------
// O16[row, h*32+v] = (sum_d sigQ*KV) / (sum_d sigQ*Ksum + 1e-6), emitted as f16
// (sole consumer is the merge GEMM's A operand).
// ---------------------------------------------------------------------------
__global__ __launch_bounds__(256)
void attn_out_kernel(const float* __restrict__ sigQ, const float* __restrict__ KV,
                     const float* __restrict__ Ksum, _Float16* __restrict__ O16)
{
  __shared__ float q[D_MODEL];
  const int row = blockIdx.x;
  const int b   = (row >= SEQ) ? 1 : 0;
  const int tid = threadIdx.x;
  const int h   = tid >> 5, v = tid & 31;

  q[tid] = sigQ[(size_t)row * D_MODEL + tid];
  __syncthreads();

  const float* qh   = &q[h * HDIM];
  const float* ksum = Ksum + (size_t)(b * NHEAD + h) * HDIM;
  const float* kv   = KV + (size_t)(b * NHEAD + h) * (HDIM * HDIM) + v;

  float z = 0.f, o = 0.f;
#pragma unroll
  for (int dd = 0; dd < HDIM; ++dd) {
    float qd = qh[dd];
    z += qd * ksum[dd];
    o += qd * kv[dd * 32];
  }
  O16[(size_t)row * D_MODEL + tid] = (_Float16)(o / (z + 1e-6f));
}

// ---------------------------------------------------------------------------
// LayerNorm over 256 with optional residual; writes f32 and/or f16 outputs.
// One wave32 per row; shfl_xor reductions.
// ---------------------------------------------------------------------------
__global__ __launch_bounds__(256)
void layernorm_kernel(const float* __restrict__ src, const float* __restrict__ g,
                      const float* __restrict__ bta, const float* __restrict__ res,
                      float* __restrict__ dstf, _Float16* __restrict__ dsth)
{
  const int lane = threadIdx.x & 31;
  const int w    = threadIdx.x >> 5;
  const size_t row = (size_t)blockIdx.x * 8 + w;
  const float* sr = src + row * D_MODEL;

  float vals[8];
  float s = 0.f;
#pragma unroll
  for (int j = 0; j < 8; ++j) { vals[j] = sr[j * 32 + lane]; s += vals[j]; }
#pragma unroll
  for (int m = 16; m >= 1; m >>= 1) s += __shfl_xor(s, m, 32);
  const float mean = s * (1.0f / 256.0f);

  float vs = 0.f;
#pragma unroll
  for (int j = 0; j < 8; ++j) { float d = vals[j] - mean; vs += d * d; }
#pragma unroll
  for (int m = 16; m >= 1; m >>= 1) vs += __shfl_xor(vs, m, 32);
  const float rs = rsqrtf(vs * (1.0f / 256.0f) + 1e-5f);

#pragma unroll
  for (int j = 0; j < 8; ++j) {
    int c = j * 32 + lane;
    float o = (vals[j] - mean) * rs * g[c] + bta[c];
    if (res) o += res[row * D_MODEL + c];
    if (dstf) dstf[row * D_MODEL + c] = o;
    if (dsth) dsth[row * D_MODEL + c] = (_Float16)o;
  }
}

// copy input feature, producing f32 working copy + f16 shadow
__global__ void copy_convert_kernel(const float* __restrict__ in,
                                    float* __restrict__ outf,
                                    _Float16* __restrict__ outh, int n)
{
  int i = blockIdx.x * blockDim.x + threadIdx.x;
  int stride = gridDim.x * blockDim.x;
  for (; i < n; i += stride) { float v = in[i]; outf[i] = v; outh[i] = (_Float16)v; }
}

// WT[n*K + k] = (f16) W[k*N + n]  — one-time weight transpose+convert
__global__ void wtrans_kernel(const float* __restrict__ W, _Float16* __restrict__ WT,
                              int K, int N)
{
  int total = K * N;
  int i = blockIdx.x * blockDim.x + threadIdx.x;
  int stride = gridDim.x * blockDim.x;
  for (; i < total; i += stride) {
    int n = i / K, k = i - n * K;
    WT[i] = (_Float16)W[(size_t)k * N + n];
  }
}

// ---------------------------------------------------------------------------
extern "C" void kernel_launch(void* const* d_in, const int* in_sizes, int n_in,
                              void* d_out, int out_size, void* d_ws, size_t ws_size,
                              hipStream_t stream)
{
  const float* f0in = (const float*)d_in[0];
  const float* f1in = (const float*)d_in[1];
  const float* Wq = (const float*)d_in[2];
  const float* Wk = (const float*)d_in[3];
  const float* Wv = (const float*)d_in[4];
  const float* Wm = (const float*)d_in[5];
  const float* W1 = (const float*)d_in[6];
  const float* W2 = (const float*)d_in[7];
  const float* g1 = (const float*)d_in[8];
  const float* b1 = (const float*)d_in[9];
  const float* g2 = (const float*)d_in[10];
  const float* b2 = (const float*)d_in[11];

  float* out = (float*)d_out;
  float* f0 = out;                         // feat0 evolves in-place in d_out
  float* f1 = out + FEAT_ELEMS;

  float* ws   = (float*)d_ws;
  float* bufQ = ws;                        // sigmaQ
  float* bufK = ws + FEAT_ELEMS;           // sigmaK / msgRaw
  float* bufV = ws + 2 * FEAT_ELEMS;       // V / mlpO
  float* KV   = ws + 3 * FEAT_ELEMS;       // [B,H,32,32]
  float* Ksum = KV + (size_t)BATCH * NHEAD * HDIM * HDIM;

  _Float16* hbase  = (_Float16*)(Ksum + BATCH * NHEAD * HDIM);
  _Float16* f0h    = hbase;                          // f16 shadow of feat0
  _Float16* f1h    = f0h + FEAT_ELEMS;
  _Float16* attnOh = f1h + FEAT_ELEMS;
  _Float16* msgNh  = attnOh + FEAT_ELEMS;
  _Float16* hidh   = msgNh + FEAT_ELEMS;             // 9600 x 512
  _Float16* WqT    = hidh + 2 * FEAT_ELEMS;          // pre-transposed f16 weights
  _Float16* WkT    = WqT + (size_t)8 * 256 * 256;
  _Float16* WvT    = WkT + (size_t)8 * 256 * 256;
  _Float16* WmT    = WvT + (size_t)8 * 256 * 256;
  _Float16* W1T    = WmT + (size_t)8 * 256 * 256;
  _Float16* W2T    = W1T + (size_t)8 * 512 * 512;

  // one-time weight transpose + convert (reused by all 16 encoder calls)
  for (int i = 0; i < 8; ++i) {
    wtrans_kernel<<<(65536 + 255) / 256, 256, 0, stream>>>(Wq + (size_t)i * 65536, WqT + (size_t)i * 65536, 256, 256);
    wtrans_kernel<<<(65536 + 255) / 256, 256, 0, stream>>>(Wk + (size_t)i * 65536, WkT + (size_t)i * 65536, 256, 256);
    wtrans_kernel<<<(65536 + 255) / 256, 256, 0, stream>>>(Wv + (size_t)i * 65536, WvT + (size_t)i * 65536, 256, 256);
    wtrans_kernel<<<(65536 + 255) / 256, 256, 0, stream>>>(Wm + (size_t)i * 65536, WmT + (size_t)i * 65536, 256, 256);
    wtrans_kernel<<<(262144 + 255) / 256, 256, 0, stream>>>(W1 + (size_t)i * 262144, W1T + (size_t)i * 262144, 512, 512);
    wtrans_kernel<<<(131072 + 255) / 256, 256, 0, stream>>>(W2 + (size_t)i * 131072, W2T + (size_t)i * 131072, 512, 256);
  }

  copy_convert_kernel<<<1024, 256, 0, stream>>>(f0in, f0, f0h, (int)FEAT_ELEMS);
  copy_convert_kernel<<<1024, 256, 0, stream>>>(f1in, f1, f1h, (int)FEAT_ELEMS);

  const dim3 blk(256);
  const dim3 g256(D_MODEL / 64, MROWS / 128);  // (4, 75)
  const dim3 g512(512 / 64,     MROWS / 128);  // (8, 75)

  auto encoder = [&](float* xf, _Float16* xh, const _Float16* srch, int i) {
    const _Float16* WqT_i = WqT + (size_t)i * 65536;
    const _Float16* WkT_i = WkT + (size_t)i * 65536;
    const _Float16* WvT_i = WvT + (size_t)i * 65536;
    const _Float16* WmT_i = WmT + (size_t)i * 65536;
    const _Float16* W1T_i = W1T + (size_t)i * 262144;
    const _Float16* W2T_i = W2T + (size_t)i * 131072;
    const float* g1_i = g1 + (size_t)i * D_MODEL;
    const float* b1_i = b1 + (size_t)i * D_MODEL;
    const float* g2_i = g2 + (size_t)i * D_MODEL;
    const float* b2_i = b2 + (size_t)i * D_MODEL;

    // sigmaQ = elu(x@Wq)+1 ; sigmaK = elu(src@Wk)+1 ; V = src@Wv   (f32 outs)
    gemm_wmma_f16<EPI_ELU1, false><<<g256, blk, 0, stream>>>(xh,   nullptr, 256, 0, WqT_i, 256, 256, bufQ, nullptr);
    gemm_wmma_f16<EPI_ELU1, false><<<g256, blk, 0, stream>>>(srch, nullptr, 256, 0, WkT_i, 256, 256, bufK, nullptr);
    gemm_wmma_f16<EPI_NONE, false><<<g256, blk, 0, stream>>>(srch, nullptr, 256, 0, WvT_i, 256, 256, bufV, nullptr);
    // per-head reductions, then O = (sigQ @ KV) * Z  (emitted as f16 shadow)
    kv_reduce_kernel<<<BATCH * NHEAD, blk, 0, stream>>>(bufK, bufV, KV, Ksum);
    attn_out_kernel<<<MROWS, blk, 0, stream>>>(bufQ, KV, Ksum, attnOh);
    // msg = LN1(attnO @ Wm)  -> f16 only (consumed by W1 GEMM)
    gemm_wmma_f16<EPI_NONE, false><<<g256, blk, 0, stream>>>(attnOh, nullptr, 256, 0, WmT_i, 256, 256, bufK, nullptr);
    layernorm_kernel<<<MROWS / 8, blk, 0, stream>>>(bufK, g1_i, b1_i, nullptr, nullptr, msgNh);
    // hid = relu([x | msg] @ W1) (f16 only) ; mlpO = hid @ W2 ; x += LN2(mlpO)
    gemm_wmma_f16<EPI_RELU, true ><<<g512, blk, 0, stream>>>(xh, msgNh, 256, 256, W1T_i, 512, 512, nullptr, hidh);
    gemm_wmma_f16<EPI_NONE, false><<<g256, blk, 0, stream>>>(hidh, nullptr, 512, 0, W2T_i, 512, 256, bufV, nullptr);
    layernorm_kernel<<<MROWS / 8, blk, 0, stream>>>(bufV, g2_i, b2_i, xf, xf, xh);
  };

  for (int i = 0; i < 8; ++i) {
    if ((i & 1) == 0) {                  // 'self'
      encoder(f0, f0h, f0h, i);
      encoder(f1, f1h, f1h, i);
    } else {                             // 'cross' (feat1 uses UPDATED feat0)
      encoder(f0, f0h, f1h, i);
      encoder(f1, f1h, f0h, i);
    }
  }
}
